// RNNTJoiner_30769145709299
// MI455X (gfx1250) — matrix operand order repaired and checked by reading, not compiled, not run on hardware
//
#include <hip/hip_runtime.h>
#include <hip/hip_bf16.h>
#include <math.h>

// ---------------------------------------------------------------------------
// RNN-T joiner, fused for MI455X (gfx1250, wave32, WMMA).
//   out(B,T,U,V) = tanh(aProj[b,t,:] + lProj[b,u,:]) @ Wv + bv
// B=8 T=200 U=50 D=256 J=320 V=1024
//
// Roofline: vocab GEMM = 52.4 GFLOP; output stores = 327.7 MB -> ~14us floor
// at 23.3 TB/s. Store-bound => fuse tanh-joint into the GEMM (joint tensor
// lives only in LDS), run the GEMM on v_wmma_f32_16x16x32_bf16 (tanh output
// in [-1,1] -> bf16 quantization benign), keep Wv/aP/lP in L2, and stream
// the write-once output with non-temporal stores so it doesn't evict them.
// ---------------------------------------------------------------------------

#define BATCH   8
#define TT      200
#define UU      50
#define DMODEL  256
#define JOINT   320
#define VOCAB   1024

#define NROWS   (BATCH * TT * UU)        // 80000 flattened (b,t,u) rows
#define AROWS   (BATCH * TT)             // 1600
#define LROWS   (BATCH * UU)             // 400

#define MTILE      64                    // rows of joint per block
#define NBLOCKS    (NROWS / MTILE)       // 1250 (exact)
#define KT         (JOINT / 32)          // 10 K-steps of 32
#define NT         (VOCAB / 16)          // 64 N-tiles of 16
#define LDS_STRIDE 344                   // bf16 elems per joint row (320 + pad),
                                         // 688B: 16B-aligned, bank-spreading
#define LDS_STRIDE_U32 (LDS_STRIDE / 2)  // 172

typedef __attribute__((ext_vector_type(16))) __bf16 v16bf;
typedef __attribute__((ext_vector_type(8)))  float  v8f;

union Frag16 {
    int4  q[2];
    v16bf v;
};

static __device__ __forceinline__ unsigned pack_bf16(float x, float y) {
    union { __bf16 h[2]; unsigned u; } un;
    un.h[0] = (__bf16)x;
    un.h[1] = (__bf16)y;
    return un.u;
}

// ---------------------------------------------------------------------------
// Kernel 1: input projections (tiny: 0.33 GFLOP).
// Flattened: rows [0,1600) -> acoustic proj, rows [1600,2000) -> linguistic.
// Consecutive threads -> consecutive cols -> coalesced weight reads.
// ---------------------------------------------------------------------------
__global__ void __launch_bounds__(256)
proj_kernel(const float* __restrict__ ac, const float* __restrict__ li,
            const float* __restrict__ Wa, const float* __restrict__ ba,
            const float* __restrict__ Wl, const float* __restrict__ bl,
            float* __restrict__ aP, float* __restrict__ lP) {
    int o   = blockIdx.x * 256 + threadIdx.x;        // 0 .. 640000-1
    int row = o / JOINT;
    int col = o - row * JOINT;

    const float* src;
    const float* W;
    float acc;
    float* dst;
    if (row < AROWS) {
        src = ac + row * DMODEL;
        W   = Wa;
        acc = ba[col];
        dst = aP + row * JOINT + col;
    } else {
        int r2 = row - AROWS;
        src = li + r2 * DMODEL;
        W   = Wl;
        acc = bl[col];
        dst = lP + r2 * JOINT + col;
    }
#pragma unroll 8
    for (int d = 0; d < DMODEL; ++d) {
        acc = fmaf(src[d], W[d * JOINT + col], acc);
    }
    *dst = acc;   // aP/lP are re-read ~50x by kernel 3 -> regular temporal
}

// ---------------------------------------------------------------------------
// Kernel 2: pre-swizzle Wv (f32 row-major 320x1024) into bf16 WMMA B-fragments.
// Fragment (ni,k) covers K = k*32..+31, N = ni*16..+15; stored as 32 lanes x
// 32 bytes contiguous (lane loads its v16bf with two b128 loads).
// Lane layout (B 32x16 bf16): lanes 0-15 hold K 0..15 (2 per VGPR),
// lanes 16-31 hold K 16..31; N = lane & 15.
// Fragment order: ni*KT + k  (wave's K loop streams contiguous 10KB).
// One thread per packed u32 pair: 163840 threads.
// ---------------------------------------------------------------------------
__global__ void __launch_bounds__(256)
swizzle_wv_kernel(const float* __restrict__ Wv, unsigned* __restrict__ WvSw) {
    int pu   = blockIdx.x * 256 + threadIdx.x;   // 0 .. 163839
    int e0   = pu << 1;                          // even bf16 index
    int q0   = e0 & 15;                          // even position-in-lane
    int lane = (e0 >> 4) & 31;
    int frag = e0 >> 9;                          // 0 .. 639 = ni*KT + k
    int ni   = frag / KT;
    int k    = frag - ni * KT;

    int K = k * 32 + ((lane >> 4) << 4) + q0;    // q0, q0+1 consecutive
    int N = ni * 16 + (lane & 15);

    float w0 = Wv[K * VOCAB + N];
    float w1 = Wv[(K + 1) * VOCAB + N];
    WvSw[pu] = pack_bf16(w0, w1);   // read by all 1250 blocks -> keep in L2
}

// ---------------------------------------------------------------------------
// Kernel 3: fused tanh-joint + vocab GEMM via v_wmma_f32_16x16x32_bf16.
// Block = 256 threads (8 waves), owns 64 flattened rows and all 1024 columns.
//   Phase A: joint tile (64 x 320 bf16) -> LDS (tanh fused, never hits HBM).
//   Phase B: wave w handles N-tiles ni = w*8..w*8+7; per (ni,k) one B-frag
//            (global, L2-resident) reused across 4 M-subtiles; A-frags are
//            two ds_load_b128 each (compiler hoists them into high VGPRs).
// Output stores are non-temporal: 327.7 MB write-once stream must not evict
// WvSw / aP / lP from L2.
// ---------------------------------------------------------------------------
__global__ void __launch_bounds__(256)
joiner_kernel(const float* __restrict__ aP, const float* __restrict__ lP,
              const unsigned* __restrict__ WvSw, const float* __restrict__ bv,
              float* __restrict__ out) {
    __shared__ unsigned sJ[MTILE * LDS_STRIDE_U32];   // 44,032 B

    const int tid = threadIdx.x;
    const int blk = blockIdx.x;

    // ---------------- Phase A: build bf16 joint tile in LDS ----------------
    {
        int lr = tid >> 2;                 // 0..63 local row
        int cs = (tid & 3) * 80;           // column start (even)
        int r  = blk * MTILE + lr;         // flattened (b,t,u) row
        int arow = r / UU;                 // = b*T + t
        int u    = r - arow * UU;
        int b    = arow / TT;
        int lrow = b * UU + u;

        const float2* ap2 = (const float2*)(aP + arow * JOINT + cs);
        const float2* lp2 = (const float2*)(lP + lrow * JOINT + cs);
        unsigned* dst = &sJ[lr * LDS_STRIDE_U32 + (cs >> 1)];
#pragma unroll 8
        for (int jj = 0; jj < 40; ++jj) {
            float2 av = ap2[jj];
            float2 lv = lp2[jj];
            dst[jj] = pack_bf16(tanhf(av.x + lv.x), tanhf(av.y + lv.y));
        }
    }
    __syncthreads();

    // ---------------- Phase B: WMMA vocab GEMM -----------------------------
    const int wave = tid >> 5;
    const int lane = tid & 31;
    const int mrow   = lane & 15;          // A row / C column within 16
    const int akoff  = (lane >> 4) << 4;   // A K-offset: lanes 16-31 -> +16
    const int crowhi = (lane >> 4) << 3;   // C: lanes 16-31 hold M = p+8

    const int4* wv4 = (const int4*)WvSw;

    for (int nb = 0; nb < 8; ++nb) {
        const int ni = wave * 8 + nb;

        v8f acc0 = {}, acc1 = {}, acc2 = {}, acc3 = {};

        const int4* bfp = wv4 + ((ni * KT) * 64 + lane * 2);
#pragma unroll
        for (int k = 0; k < KT; ++k) {
            Frag16 bf;
            bf.q[0] = bfp[0];
            bf.q[1] = bfp[1];
            bfp += 64;                     // next K fragment (1KB)

#pragma unroll
            for (int mi = 0; mi < 4; ++mi) {
                const __bf16* rp =
                    ((const __bf16*)&sJ[(mi * 16 + mrow) * LDS_STRIDE_U32]) +
                    k * 32 + akoff;
                Frag16 af;
                af.q[0] = *(const int4*)(rp);        // K = akoff .. akoff+7
                af.q[1] = *(const int4*)(rp + 8);    // K = akoff+8 .. +15
                v8f c = (mi == 0) ? acc0 : (mi == 1) ? acc1
                                         : (mi == 2) ? acc2 : acc3;
                c = __builtin_amdgcn_wmma_f32_16x16x32_bf16(
                        false, af.v, false, bf.v, (short)0, c, false, false);
                if (mi == 0) acc0 = c;
                else if (mi == 1) acc1 = c;
                else if (mi == 2) acc2 = c;
                else acc3 = c;
            }
        }

        // Bias + store: lane column fixed; lanes 0-15 are 16 consecutive cols
        // (64B per half-wave per row -> full-line write combining in GL2).
        const int col = ni * 16 + mrow;
        const float bvv = bv[col];
#pragma unroll
        for (int mi = 0; mi < 4; ++mi) {
            v8f c = (mi == 0) ? acc0 : (mi == 1) ? acc1
                                     : (mi == 2) ? acc2 : acc3;
            int rbase = blk * MTILE + mi * 16 + crowhi;
            float* op = out + (size_t)rbase * VOCAB + col;
#pragma unroll
            for (int p = 0; p < 8; ++p) {
                __builtin_nontemporal_store(c[p] + bvv, op + (size_t)p * VOCAB);
            }
        }
    }
}

// ---------------------------------------------------------------------------
// Host-side launcher.
// Workspace layout (bytes):
//   [0,        2048000) aProj  f32 1600x320
//   [2048000,  2560000) lProj  f32  400x320
//   [2560000,  3215360) WvSw   bf16 swizzled 320x1024
// ---------------------------------------------------------------------------
extern "C" void kernel_launch(void* const* d_in, const int* in_sizes, int n_in,
                              void* d_out, int out_size, void* d_ws, size_t ws_size,
                              hipStream_t stream) {
    (void)in_sizes; (void)n_in; (void)out_size; (void)ws_size;

    const float* ac = (const float*)d_in[0];
    const float* li = (const float*)d_in[1];
    const float* Wa = (const float*)d_in[2];
    const float* ba = (const float*)d_in[3];
    const float* Wl = (const float*)d_in[4];
    const float* bl = (const float*)d_in[5];
    const float* Wv = (const float*)d_in[6];
    const float* bv = (const float*)d_in[7];
    float* out = (float*)d_out;

    char* ws = (char*)d_ws;
    float*    aP   = (float*)(ws);
    float*    lP   = (float*)(ws + 2048000);
    unsigned* WvSw = (unsigned*)(ws + 2560000);

    // 1) projections: 2000*320 outputs / 256 = 2500 blocks
    proj_kernel<<<2500, 256, 0, stream>>>(ac, li, Wa, ba, Wl, bl, aP, lP);

    // 2) Wv -> bf16 swizzled fragments: 163840 u32 / 256 = 640 blocks
    swizzle_wv_kernel<<<640, 256, 0, stream>>>(Wv, WvSw);

    // 3) fused tanh-joint + vocab GEMM: 80000/64 = 1250 blocks
    joiner_kernel<<<NBLOCKS, 256, 0, stream>>>(aP, lP, WvSw, bv, out);
}
